// DOPAMINEm_sw_44375602102725
// MI455X (gfx1250) — matrix operand order repaired and checked by reading, not compiled, not run on hardware
//
#include <hip/hip_runtime.h>
#include <hip/hip_bf16.h>

typedef __attribute__((ext_vector_type(16))) _Float16 v16h;
typedef __attribute__((ext_vector_type(8)))  _Float16 v8h;
typedef __attribute__((ext_vector_type(8)))  float    v8f;

#define BATCH 4
#define HH    128
#define WW    128
#define TT    8
#define NPIX  (BATCH * HH * WW)     // 65536
#define M0_MAX 3.0f
#define P2_MAX 10.0f

// ---------------------------------------------------------------------------
// Phase 1: per-pixel Gauss-Newton (300 steps). Pure VALU, register resident.
// ---------------------------------------------------------------------------
__global__ __launch_bounds__(256) void gn_kernel(const float* __restrict__ b,
                                                 const float* __restrict__ tes,
                                                 float* __restrict__ x) {
    int p = blockIdx.x * blockDim.x + threadIdx.x;
    if (p >= NPIX) return;
    int n = p / (HH * WW);
    float bb[TT], tt[TT];
#pragma unroll
    for (int t = 0; t < TT; ++t) {
        bb[t] = b[(size_t)p * TT + t];
        tt[t] = tes[n * TT + t];
    }
    float m0 = bb[0];
#pragma unroll
    for (int t = 1; t < TT; ++t) m0 = fmaxf(m0, bb[t]);
    m0 = fminf(fmaxf(m0, 0.f), M0_MAX);
    float p2 = 1.0f;
#pragma unroll 1
    for (int it = 0; it < 300; ++it) {
        float g0 = 0.f, g1 = 0.f;
#pragma unroll
        for (int t = 0; t < TT; ++t) {
            float e = __expf(-tt[t] * p2);
            float r = m0 * e - bb[t];
            g0 += e * r;
            g1 += -tt[t] * m0 * e * r;
        }
        g0 *= (1.0f / TT);
        g1 *= (1.0f / TT);
        m0 = fminf(fmaxf(m0 - 2.f * g0, 0.f), M0_MAX);
        p2 = fminf(fmaxf(p2 - 2.f * g1, 0.f), P2_MAX);
    }
    x[(size_t)p * 2 + 0] = m0;
    x[(size_t)p * 2 + 1] = p2;
}

// f32 x -> f16 activation (2 channels, NHWC)
__global__ __launch_bounds__(256) void cvt_kernel(const float* __restrict__ x,
                                                  _Float16* __restrict__ a) {
    int i = blockIdx.x * blockDim.x + threadIdx.x;
    if (i < NPIX * 2) a[i] = (_Float16)x[i];
}

// Pre-transpose weights (HWIO f32 [k][co]) -> f16 [co][kpad], zero padded.
// Done ONCE per launch; wT stays hot in L2 (192 MB) for all conv blocks.
__global__ __launch_bounds__(256) void wtrans_kernel(const float* __restrict__ w,
                                                     _Float16* __restrict__ wT,
                                                     int ktot, int co_n, int kpad,
                                                     int cop) {
    int idx = blockIdx.x * blockDim.x + threadIdx.x;
    if (idx >= cop * kpad) return;
    int co = idx / kpad;
    int k  = idx % kpad;
    _Float16 v = (_Float16)0.f;
    if (k < ktot && co < co_n) v = (_Float16)w[(size_t)k * co_n + co];
    wT[idx] = v;
}

// ---------------------------------------------------------------------------
// 3x3 SAME conv as implicit GEMM on v_wmma_f32_16x16x32_f16.
// Block = 256 threads = 8 waves; block tile = 8x16 pixels x 64 out channels.
// Each wave owns 32 pixels x 32 channels (2 M-frags x 2 N-frags -> 4 accums):
// per K-step: 2 A frags (4x ds_load_b128) + 2 B frags (2x 32B L2 load)
// feed 4 v_wmma -> 1 LDS b128 per WMMA, 0.5 weight loads per WMMA.
// Activation halo tile is staged global->LDS with the CDNA5 async DMA path
// (global_load_async_to_lds_b128 + s_wait_asynccnt); zero-pad borders via DS.
// ---------------------------------------------------------------------------
template <int CI, int CO, bool RELU, typename OutT>
__global__ __launch_bounds__(256, 2) void conv3x3_wmma(
    const _Float16* __restrict__ act, const _Float16* __restrict__ wT,
    const float* __restrict__ bias, OutT* __restrict__ out) {
    constexpr int KTOT   = 9 * CI;
    constexpr int KSTEPS = (KTOT + 31) / 32;
    constexpr int KPAD   = KSTEPS * 32;
    constexpr int CIP    = (CI == 64) ? 72 : CI;    // pad stride: 36 dwords
    __shared__ alignas(16) _Float16 sA[180 * CIP];  // 10x18 halo tile, NHWC f16

    const int n  = blockIdx.z;
    const int y0 = blockIdx.y * 8;
    const int x0 = blockIdx.x * 16;
    const int tid  = threadIdx.x;
    const int wave = tid >> 5;
    const int lane = tid & 31;
    const int wm = wave & 3;            // 32-pixel M group (0..3)
    const int n0 = (wave >> 2) * 32;    // 32-channel N group (0..1)
    const int lhi = lane >> 4;
    const int nn  = lane & 15;          // A row / B,C column

    // Stage input tile + halo into LDS (zero-padded SAME borders).
    if constexpr (CI == 64) {
        for (int idx = tid; idx < 180 * 8; idx += 256) {  // 180 px * 8 chunks
            int p8 = idx >> 3, c8 = (idx & 7) * 8;
            int c = p8 % 18, r = p8 / 18;
            int y = y0 + r - 1, x = x0 + c - 1;
            _Float16* ldst = sA + p8 * CIP + c8;
            if (y >= 0 && y < HH && x >= 0 && x < WW) {
                const _Float16* g =
                    act + (((size_t)n * HH + y) * WW + x) * 64 + c8;
                unsigned lofs = (unsigned)(size_t)ldst;  // wave-relative LDS ofs
                asm volatile("global_load_async_to_lds_b128 %0, %1, off"
                             :: "v"(lofs), "v"(g) : "memory");
            } else {
                *(v8h*)ldst = (v8h){};                   // zero-pad border
            }
        }
        asm volatile("s_wait_asynccnt 0x0" ::: "memory");
    } else {
        for (int idx = tid; idx < 180 * CI; idx += 256) {
            int ci = idx % CI;
            int p8 = idx / CI;
            int c = p8 % 18, r = p8 / 18;
            int y = y0 + r - 1, x = x0 + c - 1;
            _Float16 v = (_Float16)0.f;
            if (y >= 0 && y < HH && x >= 0 && x < WW)
                v = act[(((size_t)n * HH + y) * WW + x) * CI + ci];
            sA[idx] = v;
        }
    }
    __syncthreads();

    v8f acc[2][2] = {};   // [mt][nt]
    const _Float16* wrow = wT + (size_t)(n0 + nn) * KPAD + lhi * 8;
#pragma unroll
    for (int s = 0; s < KSTEPS; ++s) {
        // B fragments: two 8-half runs (k = 8h.., 16+8h..) each from L2-hot weights.
        v16h bf[2];
#pragma unroll
        for (int nt = 0; nt < 2; ++nt) {
            const _Float16* bp = wrow + (size_t)nt * 16 * KPAD + s * 32;
            bf[nt] = __builtin_shufflevector(*(const v8h*)(bp), *(const v8h*)(bp + 16),
                                             0, 1, 2, 3, 4, 5, 6, 7,
                                             8, 9, 10, 11, 12, 13, 14, 15);
        }

        v16h af[2];
#pragma unroll
        for (int mt = 0; mt < 2; ++mt) {
            const int pix = wm * 32 + mt * 16 + nn;
            const int pr = pix >> 4, pc = pix & 15;
            if constexpr (CI == 64) {
                // tap fixed per step; ci = (s&1)*32 + kl; two 8-half runs.
                int tap = s >> 1;
                int ky = tap / 3, kx = tap % 3;
                const _Float16* arow = sA +
                    ((pr + ky) * 18 + (pc + kx)) * CIP + (s & 1) * 32 + lhi * 8;
                v8h a0 = *(const v8h*)(arow);
                v8h a1 = *(const v8h*)(arow + 16);
                af[mt] = __builtin_shufflevector(a0, a1, 0, 1, 2, 3, 4, 5, 6, 7,
                                                 8, 9, 10, 11, 12, 13, 14, 15);
            } else {
                // CI==2: K=18 of 32; branchless clamp + select (v_cndmask).
#pragma unroll
                for (int j = 0; j < 8; ++j) {
#pragma unroll
                    for (int e = 0; e < 2; ++e) {
                        int kl = ((j < 4) ? 0 : 16) + lhi * 8 + (j & 3) * 2 + e;
                        int tap = kl >> 1;
                        if (tap > 8) tap = 8;          // clamp: stay in halo
                        int ky = tap / 3, kx = tap % 3;
                        _Float16 v =
                            sA[((pr + ky) * 18 + (pc + kx)) * CI + (kl & 1)];
                        af[mt][j * 2 + e] = (kl < KTOT) ? v : (_Float16)0.f;
                    }
                }
            }
        }
#pragma unroll
        for (int mt = 0; mt < 2; ++mt)
#pragma unroll
            for (int nt = 0; nt < 2; ++nt)
                acc[mt][nt] = __builtin_amdgcn_wmma_f32_16x16x32_f16(
                    false, af[mt], false, bf[nt], (short)0, acc[mt][nt], false,
                    false);
    }

    // C layout: lane holds column N=lane%16; vgpr r holds row r + 8*(lane/16).
#pragma unroll
    for (int nt = 0; nt < 2; ++nt) {
        const int co = n0 + nt * 16 + nn;
        if (co < CO) {
            float bv = bias[co];
#pragma unroll
            for (int mt = 0; mt < 2; ++mt) {
#pragma unroll
                for (int r = 0; r < 8; ++r) {
                    int pp = wm * 32 + mt * 16 + r + lhi * 8;
                    int yy = y0 + (pp >> 4), xx = x0 + (pp & 15);
                    float val = acc[mt][nt][r] + bv;
                    if (RELU) val = fmaxf(val, 0.f);
                    out[(((size_t)n * HH + yy) * WW + xx) * CO + co] = (OutT)val;
                }
            }
        }
    }
}

// ---------------------------------------------------------------------------
// Per-iteration fusion: Dx = rc(x - h); gx = jr(x); x = rc(x - mu*(gx+lm*(x-Dx)))
// ---------------------------------------------------------------------------
__global__ __launch_bounds__(256) void fusion_kernel(
    float* __restrict__ x, const float* __restrict__ h,
    const float* __restrict__ b, const float* __restrict__ tes,
    const float* __restrict__ mu, const float* __restrict__ lm, int iter,
    float* __restrict__ out) {
    int p = blockIdx.x * blockDim.x + threadIdx.x;
    if (p >= NPIX) return;
    int n = p / (HH * WW);
    float m0 = x[(size_t)p * 2 + 0], p2 = x[(size_t)p * 2 + 1];
    float d0 = fminf(fmaxf(m0 - h[(size_t)p * 2 + 0], 0.f), M0_MAX);
    float d1 = fminf(fmaxf(p2 - h[(size_t)p * 2 + 1], 0.f), P2_MAX);
    size_t slot0 = (size_t)(2 * iter) * NPIX * 2;
    out[slot0 + (size_t)p * 2 + 0] = d0;
    out[slot0 + (size_t)p * 2 + 1] = d1;

    float g0 = 0.f, g1 = 0.f;
#pragma unroll
    for (int t = 0; t < TT; ++t) {
        float te = tes[n * TT + t];
        float e = __expf(-te * p2);
        float r = m0 * e - b[(size_t)p * TT + t];
        g0 += e * r;
        g1 += -te * m0 * e * r;
    }
    g0 *= (1.0f / TT);
    g1 *= (1.0f / TT);
    float L = lm[0], MU = mu[iter];
    float nx0 = fminf(fmaxf(m0 - MU * (g0 + L * (m0 - d0)), 0.f), M0_MAX);
    float nx1 = fminf(fmaxf(p2 - MU * (g1 + L * (p2 - d1)), 0.f), P2_MAX);
    x[(size_t)p * 2 + 0] = nx0;
    x[(size_t)p * 2 + 1] = nx1;
    size_t slot1 = (size_t)(2 * iter + 1) * NPIX * 2;
    out[slot1 + (size_t)p * 2 + 0] = nx0;
    out[slot1 + (size_t)p * 2 + 1] = nx1;
    if (iter == 9) {
        size_t slot2 = (size_t)20 * NPIX * 2;
        out[slot2 + (size_t)p * 2 + 0] = nx0;
        out[slot2 + (size_t)p * 2 + 1] = nx1;
    }
}

extern "C" void kernel_launch(void* const* d_in, const int* in_sizes, int n_in,
                              void* d_out, int out_size, void* d_ws, size_t ws_size,
                              hipStream_t stream) {
    (void)in_sizes; (void)n_in; (void)out_size; (void)ws_size;
    const float* b   = (const float*)d_in[0];
    const float* tes = (const float*)d_in[1];
    const float* w[7];
    const float* cb[7];
    for (int i = 0; i < 7; ++i) {
        w[i]  = (const float*)d_in[2 + i];
        cb[i] = (const float*)d_in[9 + i];
    }
    const float* mu = (const float*)d_in[16];
    const float* lm = (const float*)d_in[17];
    float* out = (float*)d_out;

    char* ws = (char*)d_ws;
    float*     xbuf = (float*)(ws);                        // 512 KB
    float*     hbuf = (float*)(ws + 524288);               // 512 KB
    _Float16*  a0   = (_Float16*)(ws + 1048576);           // 256 KB (2-ch f16)
    _Float16*  actA = (_Float16*)(ws + 1310720);           // 8 MB (64-ch f16)
    _Float16*  actB = (_Float16*)(ws + 9699328);           // 8 MB
    _Float16*  wt[7];
    size_t wtoff = 18087936;
    wt[0] = (_Float16*)(ws + wtoff);          // 64 x 32   = 4 KB
    wtoff += 64 * 32 * 2;
    for (int l = 1; l <= 5; ++l) {            // 64 x 576  = 72 KB each
        wt[l] = (_Float16*)(ws + wtoff);
        wtoff += 64 * 576 * 2;
    }
    wt[6] = (_Float16*)(ws + wtoff);          // 32 x 576  = 36 KB (co padded)

    // One-time weight transposes (deterministic, every launch).
    wtrans_kernel<<<(64 * 32 + 255) / 256, 256, 0, stream>>>(w[0], wt[0], 18, 64, 32, 64);
    for (int l = 1; l <= 5; ++l)
        wtrans_kernel<<<(64 * 576 + 255) / 256, 256, 0, stream>>>(w[l], wt[l], 576, 64, 576, 64);
    wtrans_kernel<<<(32 * 576 + 255) / 256, 256, 0, stream>>>(w[6], wt[6], 576, 2, 576, 32);

    gn_kernel<<<NPIX / 256, 256, 0, stream>>>(b, tes, xbuf);

    dim3 cgrid(WW / 16, HH / 8, BATCH);
    for (int i = 0; i < 10; ++i) {
        cvt_kernel<<<(NPIX * 2) / 256, 256, 0, stream>>>(xbuf, a0);
        conv3x3_wmma<2, 64, true, _Float16><<<cgrid, 256, 0, stream>>>(a0,   wt[0], cb[0], actA);
        conv3x3_wmma<64, 64, true, _Float16><<<cgrid, 256, 0, stream>>>(actA, wt[1], cb[1], actB);
        conv3x3_wmma<64, 64, true, _Float16><<<cgrid, 256, 0, stream>>>(actB, wt[2], cb[2], actA);
        conv3x3_wmma<64, 64, true, _Float16><<<cgrid, 256, 0, stream>>>(actA, wt[3], cb[3], actB);
        conv3x3_wmma<64, 64, true, _Float16><<<cgrid, 256, 0, stream>>>(actB, wt[4], cb[4], actA);
        conv3x3_wmma<64, 64, true, _Float16><<<cgrid, 256, 0, stream>>>(actA, wt[5], cb[5], actB);
        conv3x3_wmma<64, 2, false, float><<<cgrid, 256, 0, stream>>>(actB, wt[6], cb[6], hbuf);
        fusion_kernel<<<NPIX / 256, 256, 0, stream>>>(xbuf, hbuf, b, tes, mu, lm, i, out);
    }
}
